// CPF_17111149707613
// MI455X (gfx1250) — matrix operand order repaired and checked by reading, not compiled, not run on hardware
//
#include <hip/hip_runtime.h>
#include <hip/hip_bf16.h>

// ---------- problem constants (from reference) ----------
constexpr int kB  = 48;
constexpr int kS  = 96;
constexpr int kD  = 128;   // DK
constexpr int kQ  = 103;   // Q = NQ+1
constexpr int kQp = 112;   // Q padded to 7*16 for WMMA M-tiles
constexpr int kSteps = kS - 1;      // 95
constexpr int kClose = 10;
constexpr float cA = 0.09f, cB = 0.9f, cR = 0.01f;
constexpr float cTAU = 0.3f, cGAM = 1.0f;

// ---------- WMMA types ----------
typedef __attribute__((ext_vector_type(16))) __bf16 v16bf;
typedef __attribute__((ext_vector_type(8)))  float  v8f;
union AFrag { v16bf v; uint4 q[2]; };

__device__ __forceinline__ unsigned short f2bf(float f) {
    unsigned u = __float_as_uint(f);
    unsigned r = u + 0x7FFFu + ((u >> 16) & 1u);   // RNE
    return (unsigned short)(r >> 16);
}
__device__ __forceinline__ float sigmoidf(float x) { return 1.f / (1.f + __expf(-x)); }

// wave32 butterfly sum (no barriers)
__device__ __forceinline__ float waveSum(float v) {
#pragma unroll
    for (int m = 16; m; m >>= 1) v += __shfl_xor(v, m, 32);
    return v;
}

// ---------- workspace layout (bytes, all 256B aligned) ----------
constexpr size_t OFF_FW   = 0;                      // kB*kSteps f32   (18240 -> 18432)
constexpr size_t OFF_W1F  = 18432;                  // 65536 bf16     (131072)
constexpr size_t OFF_W4F  = OFF_W1F + 131072;       // 16384 bf16     (32768)
constexpr size_t OFF_X1   = OFF_W4F + 32768;        // kB*kS*512 bf16 (4718592)
constexpr size_t OFF_LRN  = OFF_X1 + 4718592;       // kB*kS*kD f32   (2359296)

// =====================================================================
// Kernel 1: pre-swizzle W1 (128x512) into WMMA B-fragment order, bf16.
// B-frag element (lane, e) of tile (nt, ks): N = nt*16 + lane%16,
// K = ks*32 + (e&7) + (e<8 ? 0 : 16) + (lane<16 ? 0 : 8).  B[k][n] = W1[n*512+k].
// =====================================================================
__global__ void k_w1frag(const float* __restrict__ W1, unsigned short* __restrict__ w1f) {
    int idx = blockIdx.x * blockDim.x + threadIdx.x;  // 65536 total
    int e    = idx & 15;
    int lane = (idx >> 4) & 31;
    int ks   = (idx >> 9) & 15;
    int nt   = idx >> 13;
    int n = nt * 16 + (lane & 15);
    int k = ks * 32 + (e & 7) + ((e < 8) ? 0 : 16) + ((lane < 16) ? 0 : 8);
    w1f[idx] = f2bf(W1[n * 512 + k]);
}

// Same for W4[:, :128] (the h_pre part of cat4). K < 128 -> 4 K-steps.
__global__ void k_w4frag(const float* __restrict__ W4, unsigned short* __restrict__ w4f) {
    int idx = blockIdx.x * blockDim.x + threadIdx.x;  // 16384 total
    int e    = idx & 15;
    int lane = (idx >> 4) & 31;
    int ks   = (idx >> 9) & 3;
    int nt   = idx >> 11;
    int n = nt * 16 + (lane & 15);
    int k = ks * 32 + (e & 7) + ((e < 8) ? 0 : 16) + ((lane < 16) ? 0 : 8);
    w4f[idx] = f2bf(W4[n * 512 + k]);
}

// =====================================================================
// Kernel 2: build X1 = concat([E_e[e], E_k[k], a, sa], -1) in bf16.
// =====================================================================
__global__ void k_x1(const int* __restrict__ e_data, const int* __restrict__ k_data,
                     const float* __restrict__ a_data,
                     const int* __restrict__ df_data, const int* __restrict__ al_data,
                     const int* __restrict__ at_data,
                     const float* __restrict__ E_e, const float* __restrict__ E_k,
                     const float* __restrict__ E_d, const float* __restrict__ E_al,
                     const float* __restrict__ E_at,
                     unsigned short* __restrict__ x1) {
    int idx = blockIdx.x * blockDim.x + threadIdx.x;  // kB*kS*512
    int j = idx & 511;
    int r = idx >> 9;           // b*kS + t
    int seg = j >> 7, jj = j & 127;
    float v;
    if (seg == 0)      v = E_e[e_data[r] * kD + jj];
    else if (seg == 1) v = E_k[k_data[r] * kD + jj];
    else if (seg == 2) v = a_data[r];
    else v = cA * E_d[df_data[r] * kD + jj] + cB * E_al[al_data[r] * kD + jj]
           + cR * E_at[at_data[r] * kD + jj];
    x1[idx] = f2bf(v);
}

// =====================================================================
// Kernel 3: fw / min_dt precompute (top-k smallest over batch per step).
// =====================================================================
__global__ void k_fw(const int* __restrict__ it_data, const int* __restrict__ at_data,
                     const int* __restrict__ k_data,
                     const float* __restrict__ p_matrix, const float* __restrict__ Up,
                     float* __restrict__ fw) {
    int t = threadIdx.x;
    if (t >= kSteps) return;
    float dv[kB];
    for (int b = 0; b < kB; ++b) {
        float t0 = (float)(it_data[b * kS + t]     + at_data[b * kS + t]);
        float t1 = (float)(it_data[b * kS + t + 1] + at_data[b * kS + t + 1]);
        dv[b] = fabsf(t0 - t1);
    }
    bool used[kB];
    for (int b = 0; b < kB; ++b) used[b] = false;
    int   idxs[kClose];
    float acc = 0.f;
    for (int j = 0; j < kClose; ++j) {
        int best = -1; float bv = 3.4e38f;
        for (int b = 0; b < kB; ++b) {
            if (used[b]) continue;
            float v = dv[b] + 1e-6f;
            if (v < bv) { bv = v; best = b; }
        }
        used[best] = true; idxs[j] = best; acc += bv;
    }
    float min_dt = acc / (float)kClose;
    for (int b = 0; b < kB; ++b) {
        int kt = k_data[b * kS + t];
        bool near0 = false;
        for (int j = 0; j < kClose; ++j) {
            float pv = p_matrix[kt * kQ + idxs[j]] * Up[kt * kQ + idxs[j]];
            if (pv == 0.f) near0 = true;
        }
        float w = 1.f / (1.f + __expf((dv[b] - min_dt + cTAU) / cGAM));
        fw[b * kSteps + t] = near0 ? 1.f : sigmoidf(w);
    }
}

// =====================================================================
// Kernel 4: learn = X1 (4608x512) @ W1^T + b1, bf16 WMMA, f32 out.
// =====================================================================
__global__ void k_learn_gemm(const unsigned short* __restrict__ x1,
                             const unsigned short* __restrict__ w1f,
                             const float* __restrict__ b1,
                             float* __restrict__ learn) {
    int lane = threadIdx.x & 31;
    int wave = threadIdx.x >> 5;
    int tile = blockIdx.x * 8 + wave;       // 0..2303
    int mt = tile >> 3;
    int nt = tile & 7;
    int arow = mt * 16 + (lane & 15);
    v8f acc = {};
#pragma unroll 4
    for (int ks = 0; ks < 16; ++ks) {
        int kb = ks * 32 + ((lane < 16) ? 0 : 8);
        AFrag a;
        a.q[0] = *(const uint4*)&x1[arow * 512 + kb];
        a.q[1] = *(const uint4*)&x1[arow * 512 + kb + 16];
        v16bf bf = *(const v16bf*)&w1f[((nt * 16 + ks) * 32 + lane) << 4];
        acc = __builtin_amdgcn_wmma_f32_16x16x32_bf16(false, a.v, false, bf,
                                                      (short)0, acc, false, false);
    }
    int N = lane & 15;
    int Mb = (lane < 16) ? 0 : 8;
    int col = nt * 16 + N;
    float bias = b1[col];
#pragma unroll
    for (int r = 0; r < 8; ++r) {
        int row = mt * 16 + Mb + r;
        learn[row * kD + col] = acc[r] + bias;
    }
}

// =====================================================================
// Kernel 5: persistent recurrent scan. One block per batch, 8 waves,
// h state resident in LDS; per-step 112x128x128 bf16 WMMA GEMM with
// register-resident W4 fragments; all vector phases use all 256 threads.
// =====================================================================
__global__ void __launch_bounds__(256) k_scan(
        const int* __restrict__ k_data, const int* __restrict__ e_data,
        const int* __restrict__ at_data, const int* __restrict__ it_data,
        const int* __restrict__ al_data, const int* __restrict__ df_data,
        const float* __restrict__ E_at, const float* __restrict__ E_it,
        const float* __restrict__ E_e,  const float* __restrict__ E_k,
        const float* __restrict__ E_d,  const float* __restrict__ E_al,
        const float* __restrict__ E_disc,
        const float* __restrict__ q_matrix, const float* __restrict__ p_matrix,
        const float* __restrict__ Uq, const float* __restrict__ Up,
        const float* __restrict__ W2, const float* __restrict__ b2,
        const float* __restrict__ W3, const float* __restrict__ b3,
        const float* __restrict__ W4, const float* __restrict__ b4,
        const float* __restrict__ W6, const float* __restrict__ b6,
        const float* __restrict__ h0,
        const float* __restrict__ learn, const unsigned short* __restrict__ w4f,
        const float* __restrict__ fw,
        float* __restrict__ out) {
    const int b = blockIdx.x;
    const int tid = threadIdx.x;
    const int lane = tid & 31;
    const int wave = tid >> 5;
    const int dcol = tid & 127;         // column id for split-dot phases
    const int half = tid >> 7;          // 0/1: which half of the dot

    __shared__ float h[kQp * kD];            // 57344 B
    __shared__ unsigned short hbf[kQp * kD]; // 28672 B
    __shared__ float pe[kQp], qe[kQp];
    __shared__ float s_[kD], LG[kD], c4[kD], ht[kD];
    __shared__ float common[384], lgin[256], z6[512];
    __shared__ float part[256], partB[256], red[8];

    // ---- register-resident W4 B-fragments (invariant over the scan) ----
    const int nt = wave;                     // this wave's N-tile
    v16bf bfr[4];
#pragma unroll
    for (int ks = 0; ks < 4; ++ks)
        bfr[ks] = *(const v16bf*)&w4f[((nt * 4 + ks) * 32 + lane) << 4];

    // ---- init h (pad rows >= kQ to zero) ----
    for (int i = tid; i < kQp * kD; i += 256) {
        int q = i >> 7;
        h[i] = (q < kQ) ? h0[i] : 0.f;
    }
    __syncthreads();

    // ---- ht0 = (q_matrix[e0]*Uq[e0]) . h0 (split across 256 threads) ----
    {
        int e0 = e_data[b * kS];
        int q0 = half ? 52 : 0, q1 = half ? kQ : 52;
        float a = 0.f;
        for (int q = q0; q < q1; ++q)
            a += q_matrix[e0 * kQ + q] * Uq[e0 * kQ + q] * h[q * kD + dcol];
        part[tid] = a;
        if (tid == 0) out[b * kS] = 0.f;
    }
    __syncthreads();
    if (tid < kD) ht[tid] = part[tid] + part[tid + 128];
    __syncthreads();

    for (int t = 0; t < kSteps; ++t) {
        const int et = e_data[b * kS + t],     kt = k_data[b * kS + t];
        const int itt = it_data[b * kS + t],   att = at_data[b * kS + t];
        const int alt = al_data[b * kS + t],   dft = df_data[b * kS + t];
        const int en = e_data[b * kS + t + 1], kn = k_data[b * kS + t + 1];
        const int atn = at_data[b * kS + t+1], aln = al_data[b * kS + t+1];
        const int dfn = df_data[b * kS + t+1];
        const float fwv = fw[b * kSteps + t];

        // prefetch next step's learn row (global_prefetch_b8)
        if (t + 1 < kSteps)
            __builtin_prefetch(learn + (b * kS + t + 1) * kD, 0, 0);

        // phase 1: gather pe/qe rows, stage lg input [learn_t, ht_pre]
        for (int q = tid; q < kQp; q += 256) {
            pe[q] = (q < kQ) ? p_matrix[kt * kQ + q] * Up[kt * kQ + q] : 0.f;
            qe[q] = (q < kQ) ? q_matrix[et * kQ + q] * Uq[et * kQ + q] : 0.f;
        }
        lgin[tid] = (tid < kD) ? learn[(b * kS + t) * kD + tid] : ht[tid - kD];
        __syncthreads();

        // phase 2: dks partials (all 256 threads)
        {
            int q0 = half ? 52 : 0, q1 = half ? kQ : 52;
            float a = 0.f;
            for (int q = q0; q < q1; ++q) a += pe[q] * h[q * kD + dcol];
            part[tid] = a;
        }
        __syncthreads();

        // phase 3: s = sigmoid(tanh(dks)); ||s|| via wave reduction
        if (tid < kD) {
            float sg = sigmoidf(tanhf(part[tid] + part[tid + 128]));
            s_[tid] = sg;
            float w = waveSum(sg * sg);
            if (lane == 0) red[wave] = w;
        }
        // phase 3b: LG partials (all 256 threads, runs before barrier)
        {
            const float* w3r = W3 + dcol * 256 + half * 128;
            const float* w2r = W2 + dcol * 256 + half * 128;
            const float* lg  = lgin + half * 128;
            float a3 = 0.f, a2 = 0.f;
            for (int j = 0; j < 128; ++j) { a3 += lg[j] * w3r[j]; a2 += lg[j] * w2r[j]; }
            partB[tid] = a3;
            part[tid]  = a3; // placeholder; overwritten below
            part[tid]  = a2;
        }
        __syncthreads();
        {
            float nrm = fmaxf(sqrtf(red[0] + red[1] + red[2] + red[3]), 1e-12f);
            if (tid < kD) {
                s_[tid] /= nrm;
                float a3 = b3[tid] + partB[tid] + partB[tid + 128];
                float a2 = b2[tid] + part[tid]  + part[tid + 128];
                LG[tid] = sigmoidf(a3) * (tanhf(a2) + 1.f) * 0.5f;
            }
        }
        __syncthreads();

        // phase 4: broadcast part of cat4; snapshot h -> bf16 for WMMA A
        for (int j = tid; j < 384; j += 256) {
            int seg = j >> 7, jj = j & 127;
            float v;
            if (seg == 0)      v = LG[jj] * fwv;
            else if (seg == 1) v = E_it[itt * kD + jj];
            else v = cA * E_d[dft * kD + jj] + cB * E_al[alt * kD + jj]
                   + cR * E_at[att * kD + jj];
            common[j] = v;
        }
        for (int i = tid; i < kQp * kD; i += 256) hbf[i] = f2bf(h[i]);
        __syncthreads();

        // phase 5: c4[d] = b4[d] + common . W4[d, 128:512] (split dots)
        {
            const float* w4r = W4 + dcol * 512 + 128 + half * 192;
            const float* cm  = common + half * 192;
            float a = 0.f;
            for (int j = 0; j < 192; ++j) a += cm[j] * w4r[j];
            part[tid] = a;
        }
        __syncthreads();
        if (tid < kD) c4[tid] = b4[tid] + part[tid] + part[tid + 128];
        __syncthreads();

        // phase 6: WMMA  Z = h_bf16 @ W4a^T ; fused gate + state update
        {
            const int N = lane & 15;
            const int Mb = (lane < 16) ? 0 : 8;
            const int d = nt * 16 + N;
            const float c4d = c4[d], lgd = LG[d], sd = s_[d];
            for (int mt = 0; mt < 7; ++mt) {
                v8f acc = {};
#pragma unroll
                for (int ks = 0; ks < 4; ++ks) {
                    int arow = mt * 16 + (lane & 15);
                    int kb = ks * 32 + ((lane < 16) ? 0 : 8);
                    AFrag a;
                    a.q[0] = *(const uint4*)&hbf[arow * kD + kb];
                    a.q[1] = *(const uint4*)&hbf[arow * kD + kb + 16];
                    acc = __builtin_amdgcn_wmma_f32_16x16x32_bf16(false, a.v, false, bfr[ks],
                                                                  (short)0, acc, false, false);
                }
#pragma unroll
                for (int r = 0; r < 8; ++r) {
                    int q = mt * 16 + Mb + r;
                    if (q < kQ) {
                        float hp = h[q * kD + d];
                        float g = sigmoidf(acc[r] + c4d);
                        h[q * kD + d] = qe[q] * lgd + sd + hp * g;  // LG_tilde + h*gate
                    }
                }
            }
        }
        __syncthreads();

        // phase 7: ht = q_matrix[e_n] . h (split dots)
        {
            int q0 = half ? 52 : 0, q1 = half ? kQ : 52;
            float a = 0.f;
            for (int q = q0; q < q1; ++q)
                a += q_matrix[en * kQ + q] * h[q * kD + dcol];
            part[tid] = a;
        }
        __syncthreads();
        if (tid < kD) ht[tid] = part[tid] + part[tid + 128];
        __syncthreads();

        // phase 8: z6 = [E_e[en], E_k[kn], e_disc_n, ht]
        for (int j = tid; j < 512; j += 256) {
            float v;
            if (j < 128)      v = E_e[en * kD + j];
            else if (j < 256) v = E_k[kn * kD + (j - 128)];
            else if (j < 384) {
                int jj = j - 256;
                float de = E_d[dfn * kD + jj];
                float sa = cA * de + cB * E_al[aln * kD + jj] + cR * E_at[atn * kD + jj];
                v = sigmoidf(E_disc[en * kD + jj]) * (sa - de);
            } else v = ht[j - 384];
            z6[j] = v;
        }
        __syncthreads();

        // phase 9: y = mean_d sigmoid(z6 . W6[d] + b6[d]) (split dots)
        {
            const float* w6r = W6 + dcol * 512 + half * 256;
            const float* z   = z6 + half * 256;
            float a = 0.f;
            for (int j = 0; j < 256; ++j) a += z[j] * w6r[j];
            part[tid] = a;
        }
        __syncthreads();
        if (tid < kD) {
            float sv = sigmoidf(b6[tid] + part[tid] + part[tid + 128]);
            float w = waveSum(sv);
            if (lane == 0) red[wave] = w;
        }
        __syncthreads();
        if (tid == 0)
            out[b * kS + t + 1] = (red[0] + red[1] + red[2] + red[3]) / (float)kD;
        __syncthreads();
    }
}

// =====================================================================
extern "C" void kernel_launch(void* const* d_in, const int* in_sizes, int n_in,
                              void* d_out, int out_size, void* d_ws, size_t ws_size,
                              hipStream_t stream) {
    const int*   k_data  = (const int*)d_in[0];
    const int*   e_data  = (const int*)d_in[1];
    const int*   at_data = (const int*)d_in[2];
    const int*   it_data = (const int*)d_in[3];
    const int*   al_data = (const int*)d_in[4];
    const int*   df_data = (const int*)d_in[5];
    const float* a_data  = (const float*)d_in[6];
    const float* E_at    = (const float*)d_in[7];
    const float* E_it    = (const float*)d_in[8];
    const float* E_e     = (const float*)d_in[9];
    const float* E_k     = (const float*)d_in[10];
    const float* E_d     = (const float*)d_in[11];
    const float* E_al    = (const float*)d_in[12];
    const float* E_disc  = (const float*)d_in[13];
    const float* q_matrix= (const float*)d_in[14];
    const float* p_matrix= (const float*)d_in[15];
    const float* Uq      = (const float*)d_in[16];
    const float* Up      = (const float*)d_in[17];
    const float* W1      = (const float*)d_in[18];
    const float* b1      = (const float*)d_in[19];
    const float* W2      = (const float*)d_in[20];
    const float* b2      = (const float*)d_in[21];
    const float* W3      = (const float*)d_in[22];
    const float* b3      = (const float*)d_in[23];
    const float* W4      = (const float*)d_in[24];
    const float* b4      = (const float*)d_in[25];
    const float* W6      = (const float*)d_in[26];
    const float* b6      = (const float*)d_in[27];
    const float* h0      = (const float*)d_in[28];
    float* out = (float*)d_out;

    char* ws = (char*)d_ws;
    float*          fw_ws  = (float*)(ws + OFF_FW);
    unsigned short* w1f    = (unsigned short*)(ws + OFF_W1F);
    unsigned short* w4f    = (unsigned short*)(ws + OFF_W4F);
    unsigned short* x1     = (unsigned short*)(ws + OFF_X1);
    float*          learn  = (float*)(ws + OFF_LRN);

    // prep
    k_w1frag<<<256, 256, 0, stream>>>(W1, w1f);
    k_w4frag<<<64, 256, 0, stream>>>(W4, w4f);
    k_x1<<<(kB * kS * 512) / 256, 256, 0, stream>>>(
        e_data, k_data, a_data, df_data, al_data, at_data,
        E_e, E_k, E_d, E_al, E_at, x1);
    k_fw<<<1, 128, 0, stream>>>(it_data, at_data, k_data, p_matrix, Up, fw_ws);

    // all_learning via WMMA
    k_learn_gemm<<<288, 256, 0, stream>>>(x1, w1f, b1, learn);

    // recurrent scan: one block per batch element, state in LDS
    k_scan<<<kB, 256, 0, stream>>>(
        k_data, e_data, at_data, it_data, al_data, df_data,
        E_at, E_it, E_e, E_k, E_d, E_al, E_disc,
        q_matrix, p_matrix, Uq, Up,
        W2, b2, W3, b3, W4, b4, W6, b6, h0,
        learn, w4f, fw_ws, out);
}
